// GaussianEnergyWellAttention_27951647162820
// MI455X (gfx1250) — matrix operand order repaired and checked by reading, not compiled, not run on hardware
//
#include <hip/hip_runtime.h>

// ---------------- problem constants ----------------
#define B_   8
#define SQ_  2048
#define SK_  2048
#define D_   512

typedef __attribute__((ext_vector_type(16))) __bf16   v16bf;
typedef __attribute__((ext_vector_type(8)))  __bf16   v8bf;
typedef __attribute__((ext_vector_type(8)))  float    v8f;
typedef __attribute__((ext_vector_type(4)))  unsigned v4u;
typedef __attribute__((ext_vector_type(8)))  int      v8i;
typedef __attribute__((ext_vector_type(4)))  int      v4i;

// ---- kernel1 dynamic-LDS layout (dynamic base == offset 0: no static __shared__ in k1) ----
#define K1_KF32_OFF  0                               // K tile staged f32: 16 x 512 x 4B = 32768
#define K1_KB_OFF    32768                           // K tile bf16, padded rows
#define K1_KB_STRIDE 520                             // 512 + 8 pad (bank decorrelation)
#define K1_Q2_OFF    (K1_KB_OFF + 16*K1_KB_STRIDE*2) // 49408 : 64 x f32 |q|^2
#define K1_K2_OFF    (K1_Q2_OFF + 64*4)              // 49664 : 16 x f32 |k|^2
#define K1_SMEM      (K1_K2_OFF + 16*4)              // 49728 bytes

#if __has_builtin(__builtin_amdgcn_tensor_load_to_lds)
#define HAVE_TDM 1
#else
#define HAVE_TDM 0
#endif

// 2D tile DMA: global (row-major, row_stride elems of f32) -> LDS contiguous, via TDM D#.
// This toolchain exposes the 6-arg builtin: (v4u g0, v8i g1, v4i g2, v4i g3, v8i g4, i32 cpol).
__device__ __forceinline__ void tdm_load_2d(unsigned lds_off, const float* gptr,
                                            unsigned tile_d0, unsigned tile_d1,
                                            unsigned tensor_d0, unsigned tensor_d1,
                                            unsigned row_stride_elems) {
#if HAVE_TDM
  unsigned long long ga = (unsigned long long)(uintptr_t)gptr;
  v4u g0;
  g0[0] = 1u;                                                  // count=1, user descriptor
  g0[1] = lds_off;                                             // lds_addr (bytes)
  g0[2] = (unsigned)ga;                                        // global_addr[31:0]
  g0[3] = (unsigned)((ga >> 32) & 0x1FFFFFFu) | (2u << 30);    // global_addr[56:32] | type=2
  v8i g1;
  g1[0] = (int)(2u << 16);                                     // data_size=2 (4B); no multicast/pad
  g1[1] = (int)((tensor_d0 & 0xFFFFu) << 16);                  // tensor_dim0[15:0]
  g1[2] = (int)((tensor_d0 >> 16) | ((tensor_d1 & 0xFFFFu) << 16));
  g1[3] = (int)((tensor_d1 >> 16) | (tile_d0 << 16));          // tile_dim0
  g1[4] = (int)(tile_d1 & 0xFFFFu);                            // tile_dim1 (tile_dim2=0 -> 2D)
  g1[5] = (int)row_stride_elems;                               // tensor_dim0_stride[31:0]
  g1[6] = 0;
  g1[7] = 0;
  v4i g2; g2[0]=0; g2[1]=0; g2[2]=0; g2[3]=0;                  // 2D: groups 2/3 unused
  v4i g3; g3[0]=0; g3[1]=0; g3[2]=0; g3[3]=0;
  v8i g4; g4[0]=0; g4[1]=0; g4[2]=0; g4[3]=0; g4[4]=0; g4[5]=0; g4[6]=0; g4[7]=0;
  __builtin_amdgcn_tensor_load_to_lds(g0, g1, g2, g3, g4, 0);
#else
  // Fallback: one wave (32 lanes) synchronous copy.
  extern __shared__ char smem_fb[];
  float* dst = (float*)(smem_fb + lds_off);
  const int lane = threadIdx.x & 31;
  const int total = (int)(tile_d0 * tile_d1);
  for (int i = lane * 4; i < total; i += 128) {
    const int row = i / (int)tile_d0, col = i % (int)tile_d0;
    float4 v = *(const float4*)(gptr + (size_t)row * row_stride_elems + col);
    *(float4*)(dst + i) = v;
  }
#endif
}

__device__ __forceinline__ void tdm_wait() {
#if __has_builtin(__builtin_amdgcn_s_wait_tensorcnt)
#if HAVE_TDM
  __builtin_amdgcn_s_wait_tensorcnt(0);
#endif
#endif
}

// =====================================================================================
// Kernel 1: scores. Per block: 64 queries (4 waves x 16 rows). Streams K in 16-key tiles
// via TDM, computes e = exp(exp(-alpha * max(|q|^2+|k|^2-2 q.k, 0))) (unnormalized
// softmax numerator; argument bounded in [0,1] so no max-subtraction needed), writes e
// to the weights buffer and row sums to workspace.
// =====================================================================================
__global__ void __launch_bounds__(128)
k1_qk_exp(const float* __restrict__ Q, const float* __restrict__ K,
          const float* __restrict__ alphaP, float* __restrict__ Wgt,
          float* __restrict__ rowsum) {
  extern __shared__ char smem[];
  float*  ldsKf = (float*)(smem + K1_KF32_OFF);
  __bf16* ldsKb = (__bf16*)(smem + K1_KB_OFF);
  float*  ldsQ2 = (float*)(smem + K1_Q2_OFF);
  float*  ldsK2 = (float*)(smem + K1_K2_OFF);

  const int tid  = threadIdx.x;
  const int wave = tid >> 5;
  const int lane = tid & 31;
  const int r    = lane & 15;   // A-row / B-col / C-col within 16x16
  const int hi   = lane >> 4;   // half-wave select
  const int b    = blockIdx.x >> 5;
  const int qb   = (blockIdx.x & 31) * 64;
  const float alpha = *alphaP;

  // ---- Q fragments in registers (A operand layout: lanes 0-15 K {0..7,16..23}) + |q|^2 ----
  v16bf qf[16];
  float q2p = 0.f;
  const float* qrp = Q + ((size_t)b * SQ_ + qb + wave * 16 + r) * D_;
  #pragma unroll
  for (int f = 0; f < 16; ++f) {
    const int o1 = f * 32 + hi * 8;
    float4 x0 = *(const float4*)(qrp + o1);
    float4 x1 = *(const float4*)(qrp + o1 + 4);
    float4 y0 = *(const float4*)(qrp + o1 + 16);
    float4 y1 = *(const float4*)(qrp + o1 + 20);
    q2p += x0.x*x0.x + x0.y*x0.y + x0.z*x0.z + x0.w*x0.w
         + x1.x*x1.x + x1.y*x1.y + x1.z*x1.z + x1.w*x1.w
         + y0.x*y0.x + y0.y*y0.y + y0.z*y0.z + y0.w*y0.w
         + y1.x*y1.x + y1.y*y1.y + y1.z*y1.z + y1.w*y1.w;
    qf[f][0]=(__bf16)x0.x; qf[f][1]=(__bf16)x0.y; qf[f][2] =(__bf16)x0.z; qf[f][3] =(__bf16)x0.w;
    qf[f][4]=(__bf16)x1.x; qf[f][5]=(__bf16)x1.y; qf[f][6] =(__bf16)x1.z; qf[f][7] =(__bf16)x1.w;
    qf[f][8]=(__bf16)y0.x; qf[f][9]=(__bf16)y0.y; qf[f][10]=(__bf16)y0.z; qf[f][11]=(__bf16)y0.w;
    qf[f][12]=(__bf16)y1.x; qf[f][13]=(__bf16)y1.y; qf[f][14]=(__bf16)y1.z; qf[f][15]=(__bf16)y1.w;
  }
  float q2tot = q2p + __shfl_xor(q2p, 16, 32);   // combine the two d-halves of a row
  if (hi == 0) ldsQ2[wave * 16 + r] = q2tot;

  const float* Kbase = K + (size_t)b * SK_ * D_;
  if (tid < 32) tdm_load_2d(K1_KF32_OFF, Kbase, D_, 16, D_, SK_, D_);  // first tile
  __syncthreads();

  float q2m[8];
  #pragma unroll
  for (int v = 0; v < 8; ++v) q2m[v] = ldsQ2[wave * 16 + v + hi * 8];
  float rowacc[8];
  #pragma unroll
  for (int v = 0; v < 8; ++v) rowacc[v] = 0.f;

  float* Wrow = Wgt + ((size_t)b * SQ_ + qb + wave * 16) * SK_;

  for (int kt = 0; kt < SK_ / 16; ++kt) {
    if (tid < 32) tdm_wait();
    __syncthreads();

    // f32 -> bf16 conversion into padded tile + |k|^2 (8 lanes per key, shuffle-reduce)
    {
      const int kk = tid >> 3, s = tid & 7;
      const float* src = ldsKf + kk * D_;
      __bf16* dst = ldsKb + kk * K1_KB_STRIDE;
      float acc = 0.f;
      #pragma unroll
      for (int j = 0; j < 64; ++j) {
        const int d = s + j * 8;
        const float x = src[d];
        acc += x * x;
        dst[d] = (__bf16)x;
      }
      acc += __shfl_xor(acc, 1, 32);
      acc += __shfl_xor(acc, 2, 32);
      acc += __shfl_xor(acc, 4, 32);
      if (s == 0) ldsK2[kk] = acc;
    }
    __syncthreads();

    // prefetch next K tile while all waves compute (DMA only touches ldsKf)
    if (tid < 32 && kt + 1 < SK_ / 16)
      tdm_load_2d(K1_KF32_OFF, Kbase + (size_t)(kt + 1) * 16 * D_,
                  D_, 16, D_, SK_ - (kt + 1) * 16, D_);

    // S = Q . K^T : 16 chained bf16 WMMAs over D=512
    v8f S;
    #pragma unroll
    for (int i = 0; i < 8; ++i) S[i] = 0.f;
    #pragma unroll
    for (int f = 0; f < 16; ++f) {
      const __bf16* kp = ldsKb + r * K1_KB_STRIDE + f * 32 + hi * 8; // B: lane=col=key
      v8bf lo  = *(const v8bf*)kp;
      v8bf h8  = *(const v8bf*)(kp + 16);
      v16bf kf;
      #pragma unroll
      for (int i = 0; i < 8; ++i) { kf[i] = lo[i]; kf[i + 8] = h8[i]; }
      S = __builtin_amdgcn_wmma_f32_16x16x32_bf16(false, qf[f], false, kf,
                                                  (short)0, S, false, false);
    }

    // epilogue: squared distance -> Gaussian kernel -> softmax numerator
    const float k2n = ldsK2[r];
    #pragma unroll
    for (int v = 0; v < 8; ++v) {
      float dist = q2m[v] + k2n - 2.0f * S[v];
      dist = fmaxf(dist, 0.0f);
      const float kv = __expf(-alpha * dist);   // in (0,1]
      const float e  = __expf(kv);              // softmax numerator, in [1,e] -> stable
      rowacc[v] += e;
      Wrow[(size_t)(v + hi * 8) * SK_ + kt * 16 + r] = e;
    }
    __syncthreads();   // protect ldsKb / ldsK2 before next conversion phase
  }

  // reduce row sums across the 16 columns and store once per row
  #pragma unroll
  for (int v = 0; v < 8; ++v) {
    float x = rowacc[v];
    x += __shfl_xor(x, 1, 32);
    x += __shfl_xor(x, 2, 32);
    x += __shfl_xor(x, 4, 32);
    x += __shfl_xor(x, 8, 32);
    if (r == 0) rowsum[(size_t)b * SQ_ + qb + wave * 16 + v + hi * 8] = x;
  }
}

// =====================================================================================
// Kernel 2: out = softmax(kernel) @ V. Per block: 64 queries x 128 dims. A = raw e from
// the weights buffer (L2 resident), B = V tile transposed into LDS as bf16; the 1/rowsum
// normalization is row-uniform so it is folded into the store epilogue.
// =====================================================================================
__global__ void __launch_bounds__(128)
k2_pv(const float* __restrict__ Wgt, const float* __restrict__ V,
      const float* __restrict__ rowsum, float* __restrict__ out) {
  __shared__ __bf16 ldsVT[128 * 40];   // [d(128)][key(32)] stride 40 (bank decorrelation)
  __shared__ float  ldsInv[64];

  const int tid  = threadIdx.x, wave = tid >> 5, lane = tid & 31;
  const int r    = lane & 15, hi = lane >> 4;
  const int db   = (blockIdx.x & 3) * 128;
  const int qb   = ((blockIdx.x >> 2) & 31) * 64;
  const int b    = blockIdx.x >> 7;

  if (tid < 64) ldsInv[tid] = 1.0f / rowsum[(size_t)b * SQ_ + qb + tid];
  __syncthreads();
  float invm[8];
  #pragma unroll
  for (int v = 0; v < 8; ++v) invm[v] = ldsInv[wave * 16 + v + hi * 8];

  v8f acc[8];
  #pragma unroll
  for (int t = 0; t < 8; ++t) {
    #pragma unroll
    for (int i = 0; i < 8; ++i) acc[t][i] = 0.f;
  }

  const float* Wr = Wgt + ((size_t)b * SQ_ + qb + wave * 16 + r) * SK_;
  const float* Vb = V + (size_t)b * SK_ * D_ + db;

  for (int kt = 0; kt < SK_ / 32; ++kt) {
    __syncthreads();
    // stage V[32 keys x 128 dims] transposed as bf16 (coalesced global reads)
    #pragma unroll
    for (int j = 0; j < 32; ++j) {
      const int i = tid + j * 128;
      const int key = i >> 7, d = i & 127;
      const float x = Vb[(size_t)(kt * 32 + key) * D_ + d];
      ldsVT[d * 40 + key] = (__bf16)x;
    }
    __syncthreads();

    // A fragment: 16 queries x 32 keys of raw e (A layout K pattern {hi*8..+7, 16+hi*8..+7})
    const int kb1 = kt * 32 + hi * 8;
    float4 x0 = *(const float4*)(Wr + kb1);
    float4 x1 = *(const float4*)(Wr + kb1 + 4);
    float4 y0 = *(const float4*)(Wr + kb1 + 16);
    float4 y1 = *(const float4*)(Wr + kb1 + 20);
    v16bf af;
    af[0]=(__bf16)x0.x; af[1]=(__bf16)x0.y; af[2] =(__bf16)x0.z; af[3] =(__bf16)x0.w;
    af[4]=(__bf16)x1.x; af[5]=(__bf16)x1.y; af[6] =(__bf16)x1.z; af[7] =(__bf16)x1.w;
    af[8]=(__bf16)y0.x; af[9]=(__bf16)y0.y; af[10]=(__bf16)y0.z; af[11]=(__bf16)y0.w;
    af[12]=(__bf16)y1.x; af[13]=(__bf16)y1.y; af[14]=(__bf16)y1.z; af[15]=(__bf16)y1.w;

    #pragma unroll
    for (int t = 0; t < 8; ++t) {
      const __bf16* vp = ldsVT + (t * 16 + r) * 40 + hi * 8;  // B: lane=col=dim, K=keys
      v8bf lo = *(const v8bf*)vp;
      v8bf h8 = *(const v8bf*)(vp + 16);
      v16bf bfr;
      #pragma unroll
      for (int i = 0; i < 8; ++i) { bfr[i] = lo[i]; bfr[i + 8] = h8[i]; }
      acc[t] = __builtin_amdgcn_wmma_f32_16x16x32_bf16(false, af, false, bfr,
                                                       (short)0, acc[t], false, false);
    }
  }

  float* orow = out + ((size_t)b * SQ_ + qb + wave * 16) * D_ + db;
  #pragma unroll
  for (int t = 0; t < 8; ++t) {
    #pragma unroll
    for (int v = 0; v < 8; ++v)
      orow[(size_t)(v + hi * 8) * D_ + t * 16 + r] = acc[t][v] * invm[v];
  }
}

// =====================================================================================
// Kernel 3: normalize weights in place (must run after K2, which consumes raw e).
// =====================================================================================
__global__ void __launch_bounds__(256)
k3_norm(float* __restrict__ Wgt, const float* __restrict__ rowsum) {
  const size_t base = ((size_t)blockIdx.x * 256 + threadIdx.x) * 4;
  const float inv = 1.0f / rowsum[base >> 11];   // SK = 2048
  float4 w = *(float4*)(Wgt + base);
  w.x *= inv; w.y *= inv; w.z *= inv; w.w *= inv;
  *(float4*)(Wgt + base) = w;
}

extern "C" void kernel_launch(void* const* d_in, const int* in_sizes, int n_in,
                              void* d_out, int out_size, void* d_ws, size_t ws_size,
                              hipStream_t stream) {
  (void)in_sizes; (void)n_in; (void)out_size; (void)ws_size;
  const float* Q      = (const float*)d_in[0];
  const float* K      = (const float*)d_in[1];
  const float* V      = (const float*)d_in[2];
  const float* alphaP = (const float*)d_in[3];
  float* out    = (float*)d_out;
  float* Wgt    = out + (size_t)B_ * SQ_ * D_;   // weights follow out in d_out
  float* rowsum = (float*)d_ws;                  // B*SQ f32 = 64 KB scratch

  k1_qk_exp<<<dim3(B_ * (SQ_ / 64)),              dim3(128), K1_SMEM, stream>>>(Q, K, alphaP, Wgt, rowsum);
  k2_pv    <<<dim3(B_ * (SQ_ / 64) * (D_ / 128)), dim3(128), 0,       stream>>>(Wgt, V, rowsum, out);
  k3_norm  <<<dim3((unsigned)((size_t)B_ * SQ_ * SK_ / 1024)), dim3(256), 0, stream>>>(Wgt, rowsum);
}
